// BiLSTMCRF_53446573031446
// MI455X (gfx1250) — matrix operand order repaired and checked by reading, not compiled, not run on hardware
//
#include <hip/hip_runtime.h>
#include <hip/hip_bf16.h>
#include <math.h>

// ---------------------------------------------------------------------------
// BiLSTM-CRF for MI455X (gfx1250, wave32).
//  - bf16 WMMA (v_wmma_f32_16x16x32_bf16) for input GEMM, recurrent GEMM,
//    and emission GEMM; fp32 accumulation.
//  - Bulk LDS fills via GLOBAL_LOAD_ASYNC_TO_LDS_B128 (ASYNCcnt) when the
//    toolchain exposes the gfx1250 builtins; VGPR-copy fallback otherwise.
//  - LSTM recurrence: 1 workgroup per direction, Whh (128KB) + h (32KB)
//    resident in LDS, 32 waves x 32 WMMA per timestep, 2 barriers/step,
//    global_prefetch of the next step's gate tile to hide HBM latency.
//  - CRF scan: 128 single-wave blocks, wave32 shuffles (no LDS).
// Workspace use: ~174 MB (G buffer in bf16, h_cat bf16, emissions f32).
// ---------------------------------------------------------------------------

typedef __bf16 bf16_t;
typedef __attribute__((ext_vector_type(16))) __bf16 v16bf;
typedef __attribute__((ext_vector_type(8)))  __bf16 v8bf;
typedef __attribute__((ext_vector_type(8)))  float  v8f;

#define B_   128
#define T_   512
#define E_   128
#define HD_  128
#define G4_  512   // 4*Hd
#define HC_  256   // concat hidden (fwd+bwd)
#define NT_  24

#if defined(__HIP_DEVICE_COMPILE__) && __has_builtin(__builtin_amdgcn_global_load_async_to_lds_b128)
#define HAVE_ASYNC_LDS 1
#endif

// Pointer/vector types matching the async-LDS builtin signature
// (param 0: v4i in addrspace(1) == __device__; param 1: LDS side).
typedef int v4i_async __attribute__((vector_size(16)));
typedef __attribute__((address_space(1))) v4i_async* gptr_b128;
typedef __attribute__((address_space(3))) v4i_async* lptr_b128;

// 16B global -> LDS copy; async (no VGPR round trip) when available.
static __device__ __forceinline__ void async_copy_b128(void* ldst, const void* gsrc) {
#ifdef HAVE_ASYNC_LDS
  __builtin_amdgcn_global_load_async_to_lds_b128((gptr_b128)gsrc, (lptr_b128)ldst, 0, 0);
#else
  const unsigned int* s = (const unsigned int*)gsrc;
  unsigned int* d = (unsigned int*)ldst;
  d[0] = s[0]; d[1] = s[1]; d[2] = s[2]; d[3] = s[3];
#endif
}

static __device__ __forceinline__ void wait_async0() {
#ifdef HAVE_ASYNC_LDS
#if __has_builtin(__builtin_amdgcn_s_wait_asynccnt)
  __builtin_amdgcn_s_wait_asynccnt(0);
#else
  asm volatile("s_wait_asynccnt 0" ::: "memory");
#endif
#endif
}

// ---- WMMA fragment loaders (wave32; A and B both use contiguous k = 16*(lane/16) + s) --------------------

// A fragment: 16x32 bf16 tile from row-major A[m][k] (lda elems per row).
static __device__ __forceinline__ v16bf load_a16x32(const bf16_t* A, int lda,
                                                    int mbase, int kbase, int lane) {
  int m = mbase + (lane & 15);
  const bf16_t* r0 = A + (size_t)m * lda + kbase + ((lane & 16) ? 16 : 0);
  v16bf f;
  unsigned int* fw = (unsigned int*)&f;
#pragma unroll
  for (int v = 0; v < 8; ++v) fw[v] = *(const unsigned int*)(r0 + 2 * v);
  return f;
}

// B fragment: 32x16 bf16 (K x N) where B[k][n] = W[n][k], W row-major (ldk).
static __device__ __forceinline__ v16bf load_b32x16(const bf16_t* W, int ldk,
                                                    int nbase, int kbase, int lane) {
  int n = nbase + (lane & 15);
  const bf16_t* p = W + (size_t)n * ldk + kbase + ((lane & 16) ? 16 : 0);
  v16bf f;
  unsigned int* fw = (unsigned int*)&f;
#pragma unroll
  for (int v = 0; v < 8; ++v) fw[v] = *(const unsigned int*)(p + 2 * v);
  return f;
}

static __device__ __forceinline__ v8f wmma_bf16(v16bf a, v16bf b, v8f c) {
  return __builtin_amdgcn_wmma_f32_16x16x32_bf16(false, a, false, b, (short)0, c,
                                                 false, false);
}

static __device__ __forceinline__ float sigf(float x) {
  return 1.0f / (1.0f + __expf(-x));
}

// ---- Stage 0: weight conversion to bf16, bias fusion, W_out zero-pad -------

__global__ void k_convert(const float* wihf, const float* whhf,
                          const float* wihb, const float* whhb,
                          const float* wout,
                          const float* bihf, const float* bhhf,
                          const float* bihb, const float* bhhb,
                          bf16_t* wc, bf16_t* woc, float* biasc) {
  int gid = blockIdx.x * blockDim.x + threadIdx.x;
  const int NW = G4_ * E_;  // 65536 per matrix
  if (gid < 4 * NW) {
    int m = gid / NW, i = gid % NW;
    const float* src = (m == 0) ? wihf : (m == 1) ? whhf : (m == 2) ? wihb : whhb;
    wc[gid] = (bf16_t)src[i];
  } else if (gid < 4 * NW + 32 * HC_) {
    int i = gid - 4 * NW;
    int n = i / HC_, k = i % HC_;
    woc[i] = (n < NT_) ? (bf16_t)wout[n * HC_ + k] : (bf16_t)0.0f;
  } else if (gid < 4 * NW + 32 * HC_ + 2 * G4_) {
    int i = gid - 4 * NW - 32 * HC_;
    int d = i / G4_, j = i % G4_;
    biasc[i] = d ? (bihb[j] + bhhb[j]) : (bihf[j] + bhhf[j]);
  }
}

// ---- Stage 1: G[dir][t][n][b] = emb[x] @ Wih.T + bias  (bf16, [n][b] tiles) -

__global__ __launch_bounds__(1024) void k_input_gemm(const int* x, const float* emb,
                                                     const bf16_t* wc,
                                                     const float* biasc,
                                                     bf16_t* Gc) {
  __shared__ bf16_t sW[G4_ * E_];  // 128 KB
  __shared__ bf16_t sX[B_ * E_];   // 32 KB
  int t = blockIdx.x, dir = blockIdx.y;
  int tid = threadIdx.x;
  const bf16_t* wih = wc + (size_t)(dir ? 2 : 0) * (G4_ * E_);
  // 128 KB weight tile: async global->LDS, 16B per thread per iter
#pragma unroll
  for (int i = 0; i < (G4_ * E_ * 2) / (1024 * 16); ++i) {
    int byt = (tid + i * 1024) * 16;
    async_copy_b128((char*)sW + byt, (const char*)wih + byt);
  }
  // embedding gather needs f32->bf16 conversion: VALU path
#pragma unroll
  for (int i = 0; i < (B_ * E_) / 1024; ++i) {
    int idx = tid + i * 1024;
    int b = idx >> 7, e = idx & 127;
    int tok = x[(size_t)b * T_ + t];
    sX[idx] = (bf16_t)emb[(size_t)tok * E_ + e];
  }
  wait_async0();
  __syncthreads();

  int w = tid >> 5, lane = tid & 31;
  const float* bias = biasc + dir * G4_;
#pragma unroll
  for (int i = 0; i < 8; ++i) {
    int tile = w * 8 + i;             // 256 tiles: 8 Mtiles x 32 Ntiles
    int mtile = tile >> 5, ntile = tile & 31;
    int n = ntile * 16 + (lane & 15);
    float bv = bias[n];
    v8f acc;
#pragma unroll
    for (int r = 0; r < 8; ++r) acc[r] = bv;
#pragma unroll
    for (int kk = 0; kk < 4; ++kk) {
      v16bf a = load_a16x32(sX, E_, mtile * 16, kk * 32, lane);
      v16bf b = load_b32x16(sW, E_, ntile * 16, kk * 32, lane);
      acc = wmma_bf16(a, b, acc);
    }
    int m0 = mtile * 16 + ((lane & 16) ? 8 : 0);
    v8bf h8;
#pragma unroll
    for (int r = 0; r < 8; ++r) h8[r] = (bf16_t)acc[r];
    // [dir][t][n][b]: lane's 8 rows are contiguous -> one 16B store
    *(v8bf*)(Gc + (((size_t)(dir * T_ + t) * G4_) + n) * B_ + m0) = h8;
  }
}

// ---- Stage 2: LSTM recurrence (one workgroup per direction) ----------------

__global__ __launch_bounds__(1024) void k_lstm(const bf16_t* wc, const bf16_t* Gc,
                                               bf16_t* hcat) {
  __shared__ bf16_t sW[G4_ * HD_];  // 128 KB: Whh (row-major [n][k])
  __shared__ bf16_t sH[B_ * HD_];   // 32 KB:  h   (row-major [b][k])
  int dir = blockIdx.x;
  int tid = threadIdx.x;
  const bf16_t* whh = wc + (size_t)(dir ? 3 : 1) * (G4_ * E_);
#pragma unroll
  for (int i = 0; i < (G4_ * HD_ * 2) / (1024 * 16); ++i) {
    int byt = (tid + i * 1024) * 16;
    async_copy_b128((char*)sW + byt, (const char*)whh + byt);
  }
  unsigned int* hz = (unsigned int*)sH;
#pragma unroll
  for (int i = 0; i < (B_ * HD_ / 2) / 1024; ++i) hz[tid + i * 1024] = 0u;
  wait_async0();
  __syncthreads();

  int w = tid >> 5, lane = tid & 31;
  float creg[2][8];
#pragma unroll
  for (int ci = 0; ci < 2; ++ci)
#pragma unroll
    for (int r = 0; r < 8; ++r) creg[ci][r] = 0.0f;

  for (int s = 0; s < T_; ++s) {
    int t = dir ? (T_ - 1 - s) : s;
    // Prefetch next step's gate tile (accumulator init) behind this step's
    // WMMA chain: the recurrence is a 512-step serial dependency, so the
    // G-tile HBM latency must be hidden.
    if (s + 1 < T_) {
      int tn = dir ? (T_ - 2 - s) : (s + 1);
#pragma unroll
      for (int ci = 0; ci < 2; ++ci) {
        int combo = w * 2 + ci;
        int mtile = combo >> 3, htile = combo & 7;
        int m0 = mtile * 16 + ((lane & 16) ? 8 : 0);
#pragma unroll
        for (int g = 0; g < 4; ++g) {
          int n = g * HD_ + htile * 16 + (lane & 15);
          __builtin_prefetch(Gc + (((size_t)(dir * T_ + tn) * G4_) + n) * B_ + m0, 0, 3);
        }
      }
    }
    bf16_t hst[2][8];
#pragma unroll
    for (int ci = 0; ci < 2; ++ci) {
      int combo = w * 2 + ci;          // 64 combos: 8 Mtiles x 8 Htiles
      int mtile = combo >> 3, htile = combo & 7;
      v16bf a[4];
#pragma unroll
      for (int kk = 0; kk < 4; ++kk)
        a[kk] = load_a16x32(sH, HD_, mtile * 16, kk * 32, lane);
      int m0 = mtile * 16 + ((lane & 16) ? 8 : 0);
      int hd = htile * 16 + (lane & 15);
      v8f gacc[4];
#pragma unroll
      for (int g = 0; g < 4; ++g) {    // gate order: i, f, g, o
        int nb = g * HD_ + htile * 16;
        int n = nb + (lane & 15);
        v8bf g8 = *(const v8bf*)(Gc + (((size_t)(dir * T_ + t) * G4_) + n) * B_ + m0);
        v8f acc;
#pragma unroll
        for (int r = 0; r < 8; ++r) acc[r] = (float)g8[r];
#pragma unroll
        for (int kk = 0; kk < 4; ++kk) {
          v16bf bfr = load_b32x16(sW, HD_, nb, kk * 32, lane);
          acc = wmma_bf16(a[kk], bfr, acc);
        }
        gacc[g] = acc;
      }
#pragma unroll
      for (int r = 0; r < 8; ++r) {
        float ig = sigf(gacc[0][r]);
        float fg = sigf(gacc[1][r]);
        float gg = tanhf(gacc[2][r]);
        float og = sigf(gacc[3][r]);
        float c = fg * creg[ci][r] + ig * gg;
        creg[ci][r] = c;
        float h = og * tanhf(c);
        hst[ci][r] = (bf16_t)h;
        int bb = m0 + r;
        hcat[((size_t)bb * T_ + t) * HC_ + dir * HD_ + hd] = (bf16_t)h;
      }
    }
    __syncthreads();  // all reads of sH for this step are done
#pragma unroll
    for (int ci = 0; ci < 2; ++ci) {
      int combo = w * 2 + ci;
      int mtile = combo >> 3, htile = combo & 7;
      int m0 = mtile * 16 + ((lane & 16) ? 8 : 0);
      int hd = htile * 16 + (lane & 15);
#pragma unroll
      for (int r = 0; r < 8; ++r) sH[(size_t)(m0 + r) * HD_ + hd] = hst[ci][r];
    }
    __syncthreads();  // new h visible for next step
  }
}

// ---- Stage 3: emissions = h_cat @ W_out.T + b_out  (N padded 24 -> 32) -----

__global__ __launch_bounds__(256) void k_emissions(const bf16_t* hcat,
                                                   const bf16_t* woc,
                                                   const float* bout, float* em) {
  __shared__ bf16_t sA[B_ * HC_];  // 64 KB: 128 rows x 256
  __shared__ bf16_t sB[32 * HC_];  // 16 KB
  int tid = threadIdx.x;
  size_t row0 = (size_t)blockIdx.x * 128;
  const char* asrc = (const char*)(hcat + row0 * HC_);
#pragma unroll
  for (int i = 0; i < (B_ * HC_ * 2) / (256 * 16); ++i) {
    int byt = (tid + i * 256) * 16;
    async_copy_b128((char*)sA + byt, asrc + byt);
  }
#pragma unroll
  for (int i = 0; i < (32 * HC_ * 2) / (256 * 16); ++i) {
    int byt = (tid + i * 256) * 16;
    async_copy_b128((char*)sB + byt, (const char*)woc + byt);
  }
  wait_async0();
  __syncthreads();

  int w = tid >> 5, lane = tid & 31;
  int mtile = w;  // 8 waves = 8 Mtiles
#pragma unroll
  for (int ntile = 0; ntile < 2; ++ntile) {
    int n = ntile * 16 + (lane & 15);
    float bv = (n < NT_) ? bout[n] : 0.0f;
    v8f acc;
#pragma unroll
    for (int r = 0; r < 8; ++r) acc[r] = bv;
#pragma unroll
    for (int kk = 0; kk < 8; ++kk) {
      v16bf a = load_a16x32(sA, HC_, mtile * 16, kk * 32, lane);
      v16bf b = load_b32x16(sB, HC_, ntile * 16, kk * 32, lane);
      acc = wmma_bf16(a, b, acc);
    }
    if (n < NT_) {
      int m0 = mtile * 16 + ((lane & 16) ? 8 : 0);
#pragma unroll
      for (int r = 0; r < 8; ++r) {
        size_t grow = row0 + m0 + r;  // grow = b*T + t
        em[grow * NT_ + n] = acc[r];
      }
    }
  }
}

// ---- Stage 4: CRF forward scan + gold score, one wave32 per batch ----------
// mask assumed device-resident as 1-byte bool.

__global__ void k_crf(const float* em, const unsigned char* mask, const int* tags,
                      const float* transitions, const float* strans,
                      const float* etrans, float* lossb) {
  int b = blockIdx.x;
  int lane = threadIdx.x;
  bool act = lane < NT_;
  float tc[NT_];  // transitions[:, lane] (column for destination tag = lane)
  if (act) {
#pragma unroll
    for (int i = 0; i < NT_; ++i) tc[i] = transitions[i * NT_ + lane];
  }
  float score = act ? (strans[lane] + em[(size_t)b * T_ * NT_ + lane]) : -3.0e38f;

  for (int t = 1; t < T_; ++t) {
    float sc[NT_];
#pragma unroll
    for (int i = 0; i < NT_; ++i) sc[i] = __shfl(score, i, 32);  // all lanes exec
    if (act) {
      float m = -3.0e38f;
#pragma unroll
      for (int i = 0; i < NT_; ++i) m = fmaxf(m, sc[i] + tc[i]);
      float s = 0.0f;
#pragma unroll
      for (int i = 0; i < NT_; ++i) s += __expf(sc[i] + tc[i] - m);
      float ns = m + __logf(s) + em[((size_t)b * T_ + t) * NT_ + lane];
      if (mask[(size_t)b * T_ + t]) score = ns;
    }
  }
  // logZ = logsumexp(score + end_trans)
  float v = act ? (score + etrans[lane]) : -3.0e38f;
  float mx = v;
#pragma unroll
  for (int o = 16; o > 0; o >>= 1) mx = fmaxf(mx, __shfl_xor(mx, o, 32));
  float ex = act ? __expf(v - mx) : 0.0f;
#pragma unroll
  for (int o = 16; o > 0; o >>= 1) ex += __shfl_xor(ex, o, 32);
  float logZ = mx + __logf(ex);

  // gold score: parallel partial sums over t
  float gp = 0.0f;
  int lenp = 0;
  for (int t = lane; t < T_; t += 32) lenp += mask[(size_t)b * T_ + t] ? 1 : 0;
  for (int t = 1 + lane; t < T_; t += 32) {
    if (mask[(size_t)b * T_ + t]) {
      int tp = tags[(size_t)b * T_ + t - 1];
      int tt = tags[(size_t)b * T_ + t];
      gp += transitions[tp * NT_ + tt] + em[((size_t)b * T_ + t) * NT_ + tt];
    }
  }
#pragma unroll
  for (int o = 16; o > 0; o >>= 1) {
    gp += __shfl_xor(gp, o, 32);
    lenp += __shfl_xor(lenp, o, 32);
  }
  if (lane == 0) {
    int t0 = tags[(size_t)b * T_];
    float gold = strans[t0] + em[(size_t)b * T_ * NT_ + t0] + gp;
    gold += etrans[tags[(size_t)b * T_ + (lenp - 1)]];
    lossb[b] = logZ - gold;
  }
}

__global__ void k_mean(const float* lossb, float* out) {
  __shared__ float s[B_];
  int t = threadIdx.x;
  s[t] = lossb[t];
  __syncthreads();
  for (int o = B_ / 2; o > 0; o >>= 1) {
    if (t < o) s[t] += s[t + o];
    __syncthreads();
  }
  if (t == 0) out[0] = s[0] / (float)B_;
}

// ---------------------------------------------------------------------------

extern "C" void kernel_launch(void* const* d_in, const int* in_sizes, int n_in,
                              void* d_out, int out_size, void* d_ws, size_t ws_size,
                              hipStream_t stream) {
  (void)in_sizes; (void)n_in; (void)out_size; (void)ws_size;
  const int*   x     = (const int*)d_in[0];
  const int*   tags  = (const int*)d_in[1];
  const unsigned char* mask = (const unsigned char*)d_in[2];  // jnp bool = 1 byte
  const float* emb   = (const float*)d_in[3];
  const float* wihf  = (const float*)d_in[4];
  const float* whhf  = (const float*)d_in[5];
  const float* bihf  = (const float*)d_in[6];
  const float* bhhf  = (const float*)d_in[7];
  const float* wihb  = (const float*)d_in[8];
  const float* whhb  = (const float*)d_in[9];
  const float* bihb  = (const float*)d_in[10];
  const float* bhhb  = (const float*)d_in[11];
  const float* wout  = (const float*)d_in[12];
  const float* bout  = (const float*)d_in[13];
  const float* trans = (const float*)d_in[14];
  const float* strn  = (const float*)d_in[15];
  const float* etrn  = (const float*)d_in[16];

  char* ws = (char*)d_ws;
  size_t off = 0;
  auto take = [&](size_t bytes) {
    char* p = ws + off;
    off += (bytes + 255) & ~(size_t)255;
    return p;
  };
  bf16_t* Gc    = (bf16_t*)take((size_t)2 * T_ * G4_ * B_ * sizeof(bf16_t)); // 128 MB
  bf16_t* hcat  = (bf16_t*)take((size_t)B_ * T_ * HC_ * sizeof(bf16_t));     // 32 MB
  float*  em    = (float*) take((size_t)B_ * T_ * NT_ * sizeof(float));      // 6 MB
  bf16_t* wc    = (bf16_t*)take((size_t)4 * G4_ * E_ * sizeof(bf16_t));
  bf16_t* woc   = (bf16_t*)take((size_t)32 * HC_ * sizeof(bf16_t));
  float*  biasc = (float*) take((size_t)2 * G4_ * sizeof(float));
  float*  lossb = (float*) take((size_t)B_ * sizeof(float));

  int total = 4 * G4_ * E_ + 32 * HC_ + 2 * G4_;
  k_convert<<<(total + 255) / 256, 256, 0, stream>>>(
      wihf, whhf, wihb, whhb, wout, bihf, bhhf, bihb, bhhb, wc, woc, biasc);
  k_input_gemm<<<dim3(T_, 2), 1024, 0, stream>>>(x, emb, wc, biasc, Gc);
  k_lstm<<<2, 1024, 0, stream>>>(wc, Gc, hcat);
  k_emissions<<<(B_ * T_) / 128, 256, 0, stream>>>(hcat, woc, bout, em);
  k_crf<<<B_, 32, 0, stream>>>(em, mask, tags, trans, strn, etrn, lossb);
  k_mean<<<1, B_, 0, stream>>>(lossb, (float*)d_out);
}
